// RelPositionMultiHeadAttention_86165633892491
// MI455X (gfx1250) — matrix operand order repaired and checked by reading, not compiled, not run on hardware
//
#include <hip/hip_runtime.h>
#include <stdint.h>

// ---------------- problem constants ----------------
#define D_MODEL 512
#define N_HEADS 8
#define HEAD_DIM 64
#define BATCH 2
#define SEQ 2048
#define PLEN 4095              // 2T-1
#define PPAD 4096              // padded (row 4095 = zeros)
#define ROWS (BATCH * SEQ)     // 4096
#define NSTEP (SEQ / 32)       // attention column steps

#define AS1 __attribute__((address_space(1)))
#define AS3 __attribute__((address_space(3)))

typedef __attribute__((ext_vector_type(8)))  float        v8f;
typedef __attribute__((ext_vector_type(16))) __bf16       v16bf;
typedef __attribute__((ext_vector_type(4)))  unsigned int v4u;
typedef __attribute__((ext_vector_type(4)))  unsigned int u32x4;
typedef __attribute__((ext_vector_type(8)))  int          i32x8;
typedef __attribute__((ext_vector_type(4)))  int          i32x4;

// ---- CDNA5 data-movement feature detection (3-level fallback) ----
#if defined(__has_builtin)
#if __has_builtin(__builtin_amdgcn_tensor_load_to_lds) && \
    __has_builtin(__builtin_amdgcn_s_wait_tensorcnt)
#define USE_TDM 1
#endif
#if __has_builtin(__builtin_amdgcn_global_load_async_to_lds_b128) && \
    __has_builtin(__builtin_amdgcn_s_wait_asynccnt)
#define USE_ASYNC 1
#endif
#endif
#ifndef USE_TDM
#define USE_TDM 0
#endif
#ifndef USE_ASYNC
#define USE_ASYNC 0
#endif

#if USE_TDM
#define NBUF 2  // double-buffered TDM pipeline
#else
#define NBUF 1
#endif

union Frag16 {
  v4u q[2];
  v16bf b;
  unsigned short us[16];
};

__device__ inline unsigned short f2bf(float f) {
  union { float f; unsigned int u; } c; c.f = f;
  unsigned int r = c.u + 0x7FFFu + ((c.u >> 16) & 1u);  // RNE
  return (unsigned short)(r >> 16);
}
__device__ inline float bf2f(unsigned short b) {
  union { unsigned int u; float f; } c; c.u = ((unsigned int)b) << 16;
  return c.f;
}

__device__ inline v8f wmma_bf16(const Frag16 &a, const Frag16 &b, v8f c) {
  // D = A(16x32 bf16) * B(32x16 bf16) + C(16x16 f32)
  return __builtin_amdgcn_wmma_f32_16x16x32_bf16(false, a.b, false, b.b,
                                                 (short)0, c, false, false);
}

#if USE_TDM
__device__ inline unsigned lds_off(const void *p) {
  return (unsigned)(unsigned long long)(AS3 const void *)p;
}
// Issue one TDM 2-D tile load: tileH rows x tileW elems (2B), row stride
// strideElems, global -> LDS (contiguous, row-major). D# per ISA ch.8.
__device__ inline void tdm_load_2d(unsigned lds_byte_off, const void *gaddr,
                                   unsigned tileW, unsigned tileH,
                                   unsigned strideElems) {
  unsigned long long ga = (unsigned long long)gaddr;
  u32x4 g0;
  g0[0] = 1u;                                   // count=1 (valid user D#)
  g0[1] = lds_byte_off;                         // lds_addr (bytes)
  g0[2] = (unsigned)(ga & 0xFFFFFFFFu);         // global_addr[31:0]
  g0[3] = (unsigned)((ga >> 32) & 0x01FFFFFFu)  // global_addr[56:32]
          | (2u << 30);                         // type = 2 ("image")
  i32x8 g1;
  g1[0] = 0x00010000;          // workgroup_mask=0, data_size=1 (2 bytes)
  g1[1] = (int)(tileW << 16);  // tensor_dim0[15:0] (bits 63:48)
  g1[2] = (int)(tileH << 16);  // tensor_dim0[31:16]=0 | tensor_dim1[15:0]
  g1[3] = (int)(tileW << 16);  // tensor_dim1[31:16]=0 | tile_dim0
  g1[4] = (int)tileH;          // tile_dim1 | tile_dim2=0
  g1[5] = (int)strideElems;    // tensor_dim0_stride[31:0]
  g1[6] = 0;
  g1[7] = 0;
  i32x4 z4 = {0, 0, 0, 0};     // groups 2/3 unused (<=2D tensor)
#if __clang_major__ >= 23
  i32x8 z8 = {0, 0, 0, 0, 0, 0, 0, 0};
  __builtin_amdgcn_tensor_load_to_lds(g0, g1, z4, z4, z8, 0);
#else
  __builtin_amdgcn_tensor_load_to_lds(g0, g1, z4, z4, 0);
#endif
}
#endif

// ---------------- fp32 -> bf16 conversion (with zero padding) ----------------
__global__ __launch_bounds__(256) void cvt_kernel(const float *__restrict__ src,
                                                  unsigned short *__restrict__ dst,
                                                  int nsrc, int ndst) {
  int i = blockIdx.x * 256 + threadIdx.x;
  if (i < ndst) dst[i] = (i < nsrc) ? f2bf(src[i]) : (unsigned short)0;
}

// ---------------- GEMM: C[m][n] = sum_k A[m][k] * W[n][k]  (y = x @ W^T) ------
template <typename OutT>
__global__ __launch_bounds__(256) void gemm_xt(const unsigned short *__restrict__ A,
                                               const unsigned short *__restrict__ W,
                                               OutT *__restrict__ C,
                                               int M, int N, int K) {
  const int lane = threadIdx.x & 31;
  const int wid  = threadIdx.x >> 5;
  const int n = lane & 15, h = lane >> 4;
  const int gw = blockIdx.x * 8 + wid;
  const int tilesN = N >> 6;
  const int m0 = (gw / tilesN) << 4;
  const int n0 = (gw % tilesN) << 6;

  v8f acc[4];
#pragma unroll
  for (int t = 0; t < 4; ++t)
#pragma unroll
    for (int r = 0; r < 8; ++r) acc[t][r] = 0.f;

  for (int kk = 0; kk < K; kk += 32) {
    Frag16 af;
    const unsigned short *ap = A + (size_t)(m0 + n) * K + kk + 8 * h;
    af.q[0] = *(const v4u *)ap;
    af.q[1] = *(const v4u *)(ap + 16);
    __builtin_prefetch(ap + 32, 0, 0);  // global_prefetch_b8 (next k-step)
#pragma unroll
    for (int t = 0; t < 4; ++t) {
      Frag16 bf_;
      const unsigned short *bp = W + (size_t)(n0 + 16 * t + n) * K + kk + 8 * h;
      bf_.q[0] = *(const v4u *)bp;
      bf_.q[1] = *(const v4u *)(bp + 16);
      acc[t] = wmma_bf16(af, bf_, acc[t]);
    }
  }
#pragma unroll
  for (int t = 0; t < 4; ++t)
#pragma unroll
    for (int r = 0; r < 8; ++r) {
      float v = acc[t][r];
      OutT *cp = C + (size_t)(m0 + r + 8 * h) * N + n0 + 16 * t + n;
      if constexpr (sizeof(OutT) == 2) *cp = (OutT)f2bf(v);
      else *cp = (OutT)v;
    }
}

// ---------------- fused rel-position flash attention ----------------
// grid: (SEQ/128, BATCH*N_HEADS); block: 256 (8 waves). Wave handles 16 rows.
// TDM double-buffer pipeline: step i issues DMA for step i+1, then
// s_wait_tensorcnt(2) guarantees (in-order completion) step i's pair landed.
__global__ __launch_bounds__(256) void attn_kernel(
    const unsigned short *__restrict__ Qb, const unsigned short *__restrict__ Kb,
    const unsigned short *__restrict__ Vb, const unsigned short *__restrict__ Pq,
    const float *__restrict__ ubias, const float *__restrict__ vbias,
    const unsigned char *__restrict__ mask, unsigned short *__restrict__ out) {
  __shared__ unsigned short Vraw[NBUF][32 * 64];  // V tile, row-major (TDM dst)
  __shared__ unsigned short Kraw[NBUF][32 * 64];  // K tile, row-major (TDM dst)
  __shared__ unsigned short Vt[64 * 40];          // V^T tile (padded stride 40)
  __shared__ unsigned short Pt[8][16 * 40];       // per-wave P repack tile

  const int tid = threadIdx.x;
  const int lane = tid & 31;
  const int wid = tid >> 5;
  const int n = lane & 15;
  const int h = lane >> 4;

  const int bh = blockIdx.y;
  const int b = bh >> 3;
  const int hd = bh & 7;
  const int h64 = hd * HEAD_DIM;
  const int t0 = blockIdx.x * 128 + wid * 16;
  const float scale = 0.125f;  // 1/sqrt(64)

  // ---- persistent Qu / Qv A-fragments (head dim split into 2 k-steps) ----
  Frag16 quF[2], qvF[2];
#pragma unroll
  for (int ks = 0; ks < 2; ++ks) {
    Frag16 q;
    const unsigned short *qp =
        Qb + (size_t)(b * SEQ + t0 + n) * D_MODEL + h64 + 32 * ks + 8 * h;
    q.q[0] = *(const v4u *)qp;
    q.q[1] = *(const v4u *)(qp + 16);
#pragma unroll
    for (int j = 0; j < 16; ++j) {
      int kd = 32 * ks + ((j < 8) ? (8 * h + j) : (16 + 8 * h + (j - 8)));
      float qf = bf2f(q.us[j]);
      quF[ks].us[j] = f2bf(qf + ubias[h64 + kd]);
      qvF[ks].us[j] = f2bf(qf + vbias[h64 + kd]);
    }
  }

  v8f O[4];
  float Mst[8], Lst[8];
#pragma unroll
  for (int a = 0; a < 4; ++a)
#pragma unroll
    for (int r = 0; r < 8; ++r) O[a][r] = 0.f;
#pragma unroll
  for (int r = 0; r < 8; ++r) { Mst[r] = -3.0e38f; Lst[r] = 0.f; }

#if USE_TDM
  // pipeline prologue: DMA step-0 tiles into buffer 0
  if (wid == 0) {
    const unsigned short *v0 = Vb + (size_t)(b * SEQ) * D_MODEL + h64;
    const unsigned short *k0 = Kb + (size_t)(b * SEQ) * D_MODEL + h64;
    tdm_load_2d(lds_off(&Vraw[0][0]), v0, 64u, 32u, (unsigned)D_MODEL);
    tdm_load_2d(lds_off(&Kraw[0][0]), k0, 64u, 32u, (unsigned)D_MODEL);
  }
#endif

  for (int it = 0; it < NSTEP; ++it) {
    const int j0 = it * 32;
    const int buf = it & (NBUF - 1);
    const int jr = tid >> 3;
    const int cc = (tid & 7) * 8;

    __syncthreads();  // previous step's LDS reads complete

#if USE_TDM
    if (wid == 0) {  // one wave drives the Tensor Data Mover for the block
      if (it + 1 < NSTEP) {
        const unsigned short *vn =
            Vb + (size_t)(b * SEQ + j0 + 32) * D_MODEL + h64;
        const unsigned short *kn =
            Kb + (size_t)(b * SEQ + j0 + 32) * D_MODEL + h64;
        tdm_load_2d(lds_off(&Vraw[buf ^ 1][0]), vn, 64u, 32u, (unsigned)D_MODEL);
        tdm_load_2d(lds_off(&Kraw[buf ^ 1][0]), kn, 64u, 32u, (unsigned)D_MODEL);
        __builtin_amdgcn_s_wait_tensorcnt(2);  // current pair done, next in flight
      } else {
        __builtin_amdgcn_s_wait_tensorcnt(0);
      }
    }
    __syncthreads();  // staged tiles visible to all waves
    v4u vv = *(const v4u *)&Vraw[buf][jr * 64 + cc];
#elif USE_ASYNC
    const unsigned short *vsrc = Vb + (size_t)(b * SEQ + j0) * D_MODEL + h64;
    const unsigned short *ksrc = Kb + (size_t)(b * SEQ + j0) * D_MODEL + h64;
    __builtin_amdgcn_global_load_async_to_lds_b128(
        (AS1 const void *)(vsrc + (size_t)jr * D_MODEL + cc),
        (AS3 void *)&Vraw[0][jr * 64 + cc], 0, 0);
    __builtin_amdgcn_global_load_async_to_lds_b128(
        (AS1 const void *)(ksrc + (size_t)jr * D_MODEL + cc),
        (AS3 void *)&Kraw[0][jr * 64 + cc], 0, 0);
    __builtin_amdgcn_s_wait_asynccnt(0);
    __syncthreads();
    v4u vv = *(const v4u *)&Vraw[0][jr * 64 + cc];
#else
    const unsigned short *vsrc = Vb + (size_t)(b * SEQ + j0) * D_MODEL + h64;
    const unsigned short *ksrc = Kb + (size_t)(b * SEQ + j0) * D_MODEL + h64;
    v4u vv = *(const v4u *)(vsrc + (size_t)jr * D_MODEL + cc);
    *(v4u *)&Kraw[0][jr * 64 + cc] =
        *(const v4u *)(ksrc + (size_t)jr * D_MODEL + cc);
#endif
    const unsigned short *kbuf = &Kraw[buf][0];
    {  // transpose V tile into Vt (element scatter, block-cooperative)
      const unsigned short *pv = (const unsigned short *)&vv;
#pragma unroll
      for (int i = 0; i < 8; ++i) Vt[(cc + i) * 40 + jr] = pv[i];
    }
    __syncthreads();  // Vt (and Kraw) ready

    // ---- AC tile: S(16x32) = Qu(16x64) @ K^T (K frags from staged LDS) ----
    v8f S[2];
#pragma unroll
    for (int a = 0; a < 2; ++a) {
#pragma unroll
      for (int r = 0; r < 8; ++r) S[a][r] = 0.f;
#pragma unroll
      for (int ks = 0; ks < 2; ++ks) {
        Frag16 kf;
        const unsigned short *kp = kbuf + (16 * a + n) * 64 + 32 * ks + 8 * h;
        kf.q[0] = *(const v4u *)kp;
        kf.q[1] = *(const v4u *)(kp + 16);
        S[a] = wmma_bf16(quF[ks], kf, S[a]);
      }
    }

    // ---- BD band: E(16x48) = Qv(16x64) @ Pslab^T ----
    const int base = (SEQ - 1) + j0 - (t0 + 15);  // in [0, 4048]
    v8f E[3];
#pragma unroll
    for (int a = 0; a < 3; ++a) {
#pragma unroll
      for (int r = 0; r < 8; ++r) E[a][r] = 0.f;
#pragma unroll
      for (int ks = 0; ks < 2; ++ks) {
        Frag16 pf;
        const unsigned short *pp =
            Pq + (size_t)(base + 16 * a + n) * D_MODEL + h64 + 32 * ks + 8 * h;
        pf.q[0] = *(const v4u *)pp;
        pf.q[1] = *(const v4u *)(pp + 16);
        E[a] = wmma_bf16(qvF[ks], pf, E[a]);
      }
    }

    // ---- rel-shift extraction in registers + mask + scale ----
    // bd[m][jloc] = E[m][jloc + 15 - m]; same half & VGPR index -> 2 shuffles.
    float sv[2][8];
#pragma unroll
    for (int a = 0; a < 2; ++a) {
#pragma unroll
      for (int r = 0; r < 8; ++r) {
        int constp = 16 * a + 15 - r - 8 * h;  // in [0,31], uniform per half
        int off = constp & 15;
        int alo = constp >> 4;  // 0 or 1
        float vlo = alo ? E[1][r] : E[0][r];
        float vhi = alo ? E[2][r] : E[1][r];
        int src = (h << 4) | ((off + n) & 15);
        float xlo = __shfl(vlo, src, 32);
        float xhi = __shfl(vhi, src, 32);
        float bd = ((off + n) & 16) ? xhi : xlo;
        float s = (S[a][r] + bd) * scale;
        int t = t0 + r + 8 * h;
        int j = j0 + 16 * a + n;
        unsigned char mk = mask[((size_t)b * SEQ + t) * SEQ + j];
        sv[a][r] = mk ? -10000.0f : s;
      }
    }

    // ---- online softmax (row reductions across the 16-lane half) ----
#pragma unroll
    for (int r = 0; r < 8; ++r) {
      float rm = fmaxf(sv[0][r], sv[1][r]);
#pragma unroll
      for (int m = 1; m <= 8; m <<= 1) rm = fmaxf(rm, __shfl_xor(rm, m, 32));
      float mnew = fmaxf(Mst[r], rm);
      float alpha = __expf(Mst[r] - mnew);
      float p0 = __expf(sv[0][r] - mnew);
      float p1 = __expf(sv[1][r] - mnew);
      float ps = p0 + p1;
#pragma unroll
      for (int m = 1; m <= 8; m <<= 1) ps += __shfl_xor(ps, m, 32);
      Lst[r] = Lst[r] * alpha + ps;
      Mst[r] = mnew;
#pragma unroll
      for (int a = 0; a < 4; ++a) O[a][r] *= alpha;
      Pt[wid][(r + 8 * h) * 40 + n] = f2bf(p0);
      Pt[wid][(r + 8 * h) * 40 + n + 16] = f2bf(p1);
    }
    __syncthreads();

    // ---- O(16x64) += P(16x32) @ Vtile(32x64) ----
    Frag16 pfr;
    {
      const unsigned short *pp = &Pt[wid][n * 40 + 8 * h];
      pfr.q[0] = *(const v4u *)pp;
      pfr.q[1] = *(const v4u *)(pp + 16);
    }
#pragma unroll
    for (int a = 0; a < 4; ++a) {
      Frag16 vf;
      const unsigned short *vp = &Vt[(16 * a + n) * 40 + 8 * h];
      vf.q[0] = *(const v4u *)vp;
      vf.q[1] = *(const v4u *)(vp + 16);
      O[a] = wmma_bf16(pfr, vf, O[a]);
    }
  }

  // ---- normalize and write pre-projection output (B,T,H,hd) -> (B,T,D) ----
#pragma unroll
  for (int a = 0; a < 4; ++a)
#pragma unroll
    for (int r = 0; r < 8; ++r) {
      float o = O[a][r] / Lst[r];
      out[(size_t)(b * SEQ + t0 + r + 8 * h) * D_MODEL + h64 + 16 * a + n] = f2bf(o);
    }
}

// ---------------- host-side launcher ----------------
extern "C" void kernel_launch(void *const *d_in, const int *in_sizes, int n_in,
                              void *d_out, int out_size, void *d_ws, size_t ws_size,
                              hipStream_t stream) {
  const float *x = (const float *)d_in[0];
  const float *pe = (const float *)d_in[1];
  const float *Wq = (const float *)d_in[2];
  const float *Wk = (const float *)d_in[3];
  const float *Wv = (const float *)d_in[4];
  const float *Wp = (const float *)d_in[5];
  const float *Wo = (const float *)d_in[6];
  const float *ub = (const float *)d_in[7];
  const float *vb = (const float *)d_in[8];
  const unsigned char *mask = (const unsigned char *)d_in[9];

  char *ws = (char *)d_ws;
  size_t off = 0;
  auto alloc = [&](size_t bytes) {
    char *p = ws + off;
    off += (bytes + 255) & ~(size_t)255;
    return p;
  };
  unsigned short *XB = (unsigned short *)alloc((size_t)ROWS * D_MODEL * 2);
  unsigned short *PB = (unsigned short *)alloc((size_t)PPAD * D_MODEL * 2);
  unsigned short *QB = (unsigned short *)alloc((size_t)ROWS * D_MODEL * 2);
  unsigned short *KB = (unsigned short *)alloc((size_t)ROWS * D_MODEL * 2);
  unsigned short *VB = (unsigned short *)alloc((size_t)ROWS * D_MODEL * 2);
  unsigned short *PQ = (unsigned short *)alloc((size_t)PPAD * D_MODEL * 2);
  unsigned short *WQB = (unsigned short *)alloc((size_t)D_MODEL * D_MODEL * 2);
  unsigned short *WKB = (unsigned short *)alloc((size_t)D_MODEL * D_MODEL * 2);
  unsigned short *WVB = (unsigned short *)alloc((size_t)D_MODEL * D_MODEL * 2);
  unsigned short *WPB = (unsigned short *)alloc((size_t)D_MODEL * D_MODEL * 2);
  unsigned short *WOB = (unsigned short *)alloc((size_t)D_MODEL * D_MODEL * 2);
  unsigned short *AT = (unsigned short *)alloc((size_t)ROWS * D_MODEL * 2);

  const int NX = ROWS * D_MODEL;
  const int NP = PPAD * D_MODEL;
  const int NW = D_MODEL * D_MODEL;
  cvt_kernel<<<(NX + 255) / 256, 256, 0, stream>>>(x, XB, NX, NX);
  cvt_kernel<<<(NP + 255) / 256, 256, 0, stream>>>(pe, PB, PLEN * D_MODEL, NP);
  cvt_kernel<<<(NW + 255) / 256, 256, 0, stream>>>(Wq, WQB, NW, NW);
  cvt_kernel<<<(NW + 255) / 256, 256, 0, stream>>>(Wk, WKB, NW, NW);
  cvt_kernel<<<(NW + 255) / 256, 256, 0, stream>>>(Wv, WVB, NW, NW);
  cvt_kernel<<<(NW + 255) / 256, 256, 0, stream>>>(Wp, WPB, NW, NW);
  cvt_kernel<<<(NW + 255) / 256, 256, 0, stream>>>(Wo, WOB, NW, NW);

  // waves = (M/16)*(N/64) = 2048; 8 waves/block -> 256 blocks
  gemm_xt<unsigned short><<<256, 256, 0, stream>>>(XB, WQB, QB, ROWS, D_MODEL, D_MODEL);
  gemm_xt<unsigned short><<<256, 256, 0, stream>>>(XB, WKB, KB, ROWS, D_MODEL, D_MODEL);
  gemm_xt<unsigned short><<<256, 256, 0, stream>>>(XB, WVB, VB, ROWS, D_MODEL, D_MODEL);
  gemm_xt<unsigned short><<<256, 256, 0, stream>>>(PB, WPB, PQ, PPAD, D_MODEL, D_MODEL);

  attn_kernel<<<dim3(SEQ / 128, BATCH * N_HEADS), 256, 0, stream>>>(
      QB, KB, VB, PQ, ub, vb, mask, AT);

  gemm_xt<float><<<256, 256, 0, stream>>>(AT, WOB, (float *)d_out, ROWS, D_MODEL, D_MODEL);
}
